// TenVAD_14551349199170
// MI455X (gfx1250) — matrix operand order repaired and checked by reading, not compiled, not run on hardware
//
#include <hip/hip_runtime.h>
#include <hip/hip_bf16.h>

typedef __attribute__((ext_vector_type(16))) __bf16 v16bf;
typedef __attribute__((ext_vector_type(8)))  float  v8f;

namespace {

constexpr int kMT   = 64;    // samples (M rows) per workgroup
constexpr int kSA   = 168;   // A-panel LDS row stride (bf16 elems), 336B = 84 dwords -> conflict-free
constexpr int kWS   = 40;    // W-slab LDS row stride (bf16 elems), 80B = 20 dwords -> conflict-free
constexpr int kGS   = 257;   // gates LDS row stride (f32)
constexpr int kWBUF = 256 * kWS;

constexpr int OFF_A  = 0;       // 64*168*2   = 21504 B  (bf16 A panel)
constexpr int OFF_W  = 21504;   // 2*256*40*2 = 40960 B  (double-buffered W k-slabs)
constexpr int OFF_H  = 62464;   // 64*64*2    = 8192  B  (bf16 h1 stash)
constexpr int OFF_WC = 70656;   // 704*4      = 2816  B  (conv weights)
constexpr int OFF_S  = 73472;   // 21504*4    = 86016 B  (f32 scratch / gates)
constexpr int kSmemBytes = 159488;

union Frag {
  v16bf v;
  uint2 d[4];
  uint4 q[2];
};

__device__ __forceinline__ float sigf(float x) { return 1.0f / (1.0f + __expf(-x)); }

template<int GID>
__device__ __forceinline__ float wfetch(int n, int kg,
    const float* wih1, const float* whh1,
    const float* wih2, const float* whh2, const float* fc1w)
{
  if (GID == 0) {                      // combined [160 x 256]: w_ih1^T | w_hh1^T | zero-pad
    if (kg < 80)  return wih1[n * 80 + kg];
    if (kg < 144) return whh1[n * 64 + (kg - 80)];
    return 0.0f;
  } else if (GID == 1) {               // combined [128 x 256]: w_ih2^T | w_hh2^T
    return (kg < 64) ? wih2[n * 64 + kg] : whh2[n * 64 + (kg - 64)];
  } else {                             // fc1 [128 x 32]
    return fc1w[n * 128 + kg];
  }
}

// GEMM: G[64 x 256] = A(LDS bf16 [64 x 32*KS]) * W(global -> bf16 LDS k-slabs)
template<int KS, int GID>
__device__ __forceinline__ void gemm_n256(__bf16* __restrict__ Abf,
                                          __bf16* __restrict__ Wsl,
                                          float*  __restrict__ G,
    const float* wih1, const float* whh1,
    const float* wih2, const float* whh2, const float* fc1w)
{
  const int t    = threadIdx.x;
  const int wave = t >> 5;
  const int lane = t & 31;
  const int mt   = wave >> 1;          // M tile 0..3
  const int ntb  = (wave & 1) * 8;     // base N tile (8 tiles per wave)
  const int lm   = lane & 15;
  const int hi   = lane >> 4;

  v8f acc[8];
#pragma unroll
  for (int j = 0; j < 8; ++j)
#pragma unroll
    for (int i = 0; i < 8; ++i) acc[j][i] = 0.0f;

  // stage k-slab 0 (thread t = row n, 32 contiguous weights -> bf16)
  {
    const int n = t;
#pragma unroll
    for (int k = 0; k < 32; ++k)
      Wsl[n * kWS + k] = (__bf16)wfetch<GID>(n, k, wih1, whh1, wih2, whh2, fc1w);
  }
  __syncthreads();

  for (int ks = 0; ks < KS; ++ks) {
    if (ks + 1 < KS) {                 // prefetch next slab into the other buffer
      const int n = t;
      __bf16* dst = Wsl + ((ks + 1) & 1) * kWBUF + n * kWS;
#pragma unroll
      for (int k = 0; k < 32; ++k)
        dst[k] = (__bf16)wfetch<GID>(n, (ks + 1) * 32 + k, wih1, whh1, wih2, whh2, fc1w);
    }
    const __bf16* wbuf = Wsl + (ks & 1) * kWBUF;

    // A fragment: lane = M row, K chunks [k0..k0+7] and [k0+16..k0+23], k0 = ks*32 + hi*8
    Frag fa;
    const __bf16* ap = Abf + (mt * 16 + lm) * kSA + ks * 32 + hi * 8;
    fa.q[0] = *(const uint4*)ap;
    fa.q[1] = *(const uint4*)(ap + 16);

#pragma unroll
    for (int j = 0; j < 8; ++j) {
      // B fragment: lane = N col, K chunks [hi*8..hi*8+7] and [16+hi*8..16+hi*8+7]
      Frag fb;
      const __bf16* bp = wbuf + ((ntb + j) * 16 + lm) * kWS + hi * 8;
      fb.d[0] = *(const uint2*)(bp + 0);
      fb.d[1] = *(const uint2*)(bp + 4);
      fb.d[2] = *(const uint2*)(bp + 16);
      fb.d[3] = *(const uint2*)(bp + 20);
      acc[j] = __builtin_amdgcn_wmma_f32_16x16x32_bf16(false, fa.v, false, fb.v,
                                                       (short)0, acc[j], false, false);
    }
    __syncthreads();
  }

  // C/D layout: VGPR r, lane -> M = r + hi*8, N = lm
#pragma unroll
  for (int j = 0; j < 8; ++j) {
    const int n = (ntb + j) * 16 + lm;
#pragma unroll
    for (int r = 0; r < 8; ++r)
      G[(mt * 16 + r + hi * 8) * kGS + n] = acc[j][r];
  }
}

// fc1 GEMM: G[64 x 32] = A[64 x 128] * fc1_w^T, one 16x16 tile per wave
__device__ __forceinline__ void gemm_fc(__bf16* __restrict__ Abf,
                                        __bf16* __restrict__ Wsl,
                                        float*  __restrict__ G,
                                        const float* fc1w)
{
  const int t    = threadIdx.x;
  const int wave = t >> 5;
  const int lane = t & 31;
  const int mt   = wave >> 1;
  const int nt   = wave & 1;
  const int lm   = lane & 15;
  const int hi   = lane >> 4;

  v8f acc;
#pragma unroll
  for (int i = 0; i < 8; ++i) acc[i] = 0.0f;

  if (t < 32) {
#pragma unroll
    for (int k = 0; k < 32; ++k)
      Wsl[t * kWS + k] = (__bf16)fc1w[t * 128 + k];
  }
  __syncthreads();

  for (int ks = 0; ks < 4; ++ks) {
    if (ks < 3 && t < 32) {
      __bf16* dst = Wsl + ((ks + 1) & 1) * kWBUF + t * kWS;
#pragma unroll
      for (int k = 0; k < 32; ++k)
        dst[k] = (__bf16)fc1w[t * 128 + (ks + 1) * 32 + k];
    }
    const __bf16* wbuf = Wsl + (ks & 1) * kWBUF;

    Frag fa;
    const __bf16* ap = Abf + (mt * 16 + lm) * kSA + ks * 32 + hi * 8;
    fa.q[0] = *(const uint4*)ap;
    fa.q[1] = *(const uint4*)(ap + 16);

    Frag fb;
    const __bf16* bp = wbuf + (nt * 16 + lm) * kWS + hi * 8;
    fb.d[0] = *(const uint2*)(bp + 0);
    fb.d[1] = *(const uint2*)(bp + 4);
    fb.d[2] = *(const uint2*)(bp + 16);
    fb.d[3] = *(const uint2*)(bp + 20);
    acc = __builtin_amdgcn_wmma_f32_16x16x32_bf16(false, fa.v, false, fb.v,
                                                  (short)0, acc, false, false);
    __syncthreads();
  }

#pragma unroll
  for (int r = 0; r < 8; ++r)
    G[(mt * 16 + r + hi * 8) * kGS + nt * 16 + lm] = acc[r];
}

} // namespace

__global__ void __launch_bounds__(256)
tenvad_fused(const float* __restrict__ x,
             const float* __restrict__ lstm1_h, const float* __restrict__ lstm1_c,
             const float* __restrict__ lstm2_h, const float* __restrict__ lstm2_c,
             const float* __restrict__ conv_dw_w, const float* __restrict__ conv_pw_w,
             const float* __restrict__ conv_pw_b,
             const float* __restrict__ sep1_dw_w, const float* __restrict__ sep1_pw_w,
             const float* __restrict__ sep1_pw_b,
             const float* __restrict__ sep2_dw_w, const float* __restrict__ sep2_pw_w,
             const float* __restrict__ sep2_pw_b,
             const float* __restrict__ w_ih1, const float* __restrict__ w_hh1,
             const float* __restrict__ b_ih1, const float* __restrict__ b_hh1,
             const float* __restrict__ w_ih2, const float* __restrict__ w_hh2,
             const float* __restrict__ b_ih2, const float* __restrict__ b_hh2,
             const float* __restrict__ fc1_w, const float* __restrict__ fc1_b,
             const float* __restrict__ fc2_w, const float* __restrict__ fc2_b,
             float* __restrict__ out, int Btot)
{
  extern __shared__ __align__(16) char smem[];
  __bf16* Abf = (__bf16*)(smem + OFF_A);
  __bf16* Wsl = (__bf16*)(smem + OFF_W);
  __bf16* Hb  = (__bf16*)(smem + OFF_H);
  float*  WC  = (float*)(smem + OFF_WC);
  float*  S   = (float*)(smem + OFF_S);
  float*  G   = S;

  const int t  = threadIdx.x;
  const int g0 = blockIdx.x * kMT;

  const size_t h1_off = (size_t)Btot;
  const size_t c1_off = h1_off + (size_t)Btot * 64;
  const size_t h2_off = c1_off + (size_t)Btot * 64;
  const size_t c2_off = h2_off + (size_t)Btot * 64;

  // ---- stage small conv weights into LDS ----
  for (int it = t; it < 681; it += 256) {
    float v;
    if      (it <   9) v = conv_dw_w[it];
    else if (it <  25) v = conv_pw_w[it - 9];
    else if (it <  41) v = conv_pw_b[it - 25];
    else if (it <  89) v = sep1_dw_w[it - 41];
    else if (it < 345) v = sep1_pw_w[it - 89];
    else if (it < 361) v = sep1_pw_b[it - 345];
    else if (it < 409) v = sep2_dw_w[it - 361];
    else if (it < 665) v = sep2_pw_w[it - 409];
    else               v = sep2_pw_b[it - 665];
    WC[it] = v;
  }
  // zero K-padding columns 144..159 of the A panel (GEMM1 pads K=144 -> 160)
  for (int it = t; it < kMT * 16; it += 256) {
    int m = it >> 4, j = it & 15;
    Abf[m * kSA + 144 + j] = (__bf16)0.0f;
  }
  __syncthreads();

  // conv scratch layout (floats within S), regions overlapped by liveness
  float* Smax = S;              // [64][20]   dead after C1
  float* Smin = S + 1280;       // [64][20]
  float* Y1   = S + 2560;       // [64][40]   dead after C0b
  float* D1   = S + 11264;      // [64][16][10]
  float* S1   = S;              // [64][16][11]  overwrites Smax/Smin/Y1
  float* D2   = S + 11264;      // [64][16][5]   overwrites D1

  // C0a: 3x3 depthwise valid conv over x[3][41] -> y1[39]
  for (int it = t; it < kMT * 39; it += 256) {
    int m = it / 39, w = it - m * 39;
    const float* xp = x + (size_t)(g0 + m) * 123;
    float acc = 0.0f;
#pragma unroll
    for (int r = 0; r < 3; ++r)
#pragma unroll
      for (int kc = 0; kc < 3; ++kc)
        acc += xp[r * 41 + w + kc] * WC[r * 3 + kc];
    Y1[m * 40 + w] = acc;
  }
  __syncthreads();

  // C0b: per-window max/min of y1 (maxpool commutes through monotone relu(a*y+b))
  for (int it = t; it < kMT * 19; it += 256) {
    int m = it / 19, tt = it - m * 19;
    float a = Y1[m * 40 + 2 * tt], b = Y1[m * 40 + 2 * tt + 1], c = Y1[m * 40 + 2 * tt + 2];
    Smax[m * 20 + tt] = fmaxf(a, fmaxf(b, c));
    Smin[m * 20 + tt] = fminf(a, fminf(b, c));
  }
  __syncthreads();

  // C1: pointwise expand + relu + maxpool + sep1 depthwise (stride 2, pad 1)
  for (int it = t; it < kMT * 160; it += 256) {
    int m = it / 160, rem = it - m * 160, c = rem / 10, u = rem - c * 10;
    float pw = WC[9 + c], pb = WC[25 + c];
    float acc = 0.0f;
#pragma unroll
    for (int k = 0; k < 3; ++k) {
      int tt = 2 * u - 1 + k;
      float p = 0.0f;
      if (tt >= 0 && tt < 19) {
        float e = (pw > 0.0f) ? Smax[m * 20 + tt] : Smin[m * 20 + tt];
        p = fmaxf(pw * e + pb, 0.0f);
      }
      acc += p * WC[41 + c * 3 + k];
    }
    D1[m * 160 + c * 10 + u] = acc;
  }
  __syncthreads();

  // C2: sep1 pointwise + relu, with right pad (u==10 -> 0)
  for (int it = t; it < kMT * 160; it += 256) {
    int m = it / 160, rem = it - m * 160, cp = rem / 10, u = rem - cp * 10;
    float acc = WC[345 + cp];
#pragma unroll
    for (int c = 0; c < 16; ++c)
      acc += D1[m * 160 + c * 10 + u] * WC[89 + cp * 16 + c];
    S1[m * 176 + cp * 11 + u] = fmaxf(acc, 0.0f);
  }
  for (int it = t; it < kMT * 16; it += 256) {
    int m = it >> 4, cp = it & 15;
    S1[m * 176 + cp * 11 + 10] = 0.0f;
  }
  __syncthreads();

  // C3a: sep2 depthwise stride 2
  for (int it = t; it < kMT * 80; it += 256) {
    int m = it / 80, rem = it - m * 80, c = rem / 5, v = rem - c * 5;
    float acc = 0.0f;
#pragma unroll
    for (int k = 0; k < 3; ++k)
      acc += S1[m * 176 + c * 11 + 2 * v + k] * WC[361 + c * 3 + k];
    D2[m * 160 + c * 5 + v] = acc;
  }
  __syncthreads();

  // C3b: sep2 pointwise + relu -> feat (feat[v*16+cp]) into A panel as bf16
  for (int it = t; it < kMT * 80; it += 256) {
    int m = it / 80, rem = it - m * 80, cp = rem / 5, v = rem - cp * 5;
    float acc = WC[665 + cp];
#pragma unroll
    for (int c = 0; c < 16; ++c)
      acc += D2[m * 160 + c * 5 + v] * WC[409 + cp * 16 + c];
    Abf[m * kSA + v * 16 + cp] = (__bf16)fmaxf(acc, 0.0f);
  }
  // A1[:,80:144] = previous h1
  for (int it = t; it < kMT * 64; it += 256) {
    int m = it >> 6, j = it & 63;
    Abf[m * kSA + 80 + j] = (__bf16)lstm1_h[(size_t)(g0 + m) * 64 + j];
  }
  __syncthreads();

  // ---- LSTM1 gates: [64 x 256] = A1[64 x 160] x W1 ----
  gemm_n256<5, 0>(Abf, Wsl, G, w_ih1, w_hh1, w_ih2, w_hh2, fc1_w);
  __syncthreads();

  // elementwise LSTM1 (gate order i,f,g,o)
  for (int it = t; it < kMT * 64; it += 256) {
    int m = it >> 6, j = it & 63;
    size_t gix = (size_t)(g0 + m) * 64 + j;
    float gi = G[m * kGS + j]       + b_ih1[j]       + b_hh1[j];
    float gf = G[m * kGS + 64 + j]  + b_ih1[64 + j]  + b_hh1[64 + j];
    float gg = G[m * kGS + 128 + j] + b_ih1[128 + j] + b_hh1[128 + j];
    float go = G[m * kGS + 192 + j] + b_ih1[192 + j] + b_hh1[192 + j];
    float cp = lstm1_c[gix];
    float cn = sigf(gf) * cp + sigf(gi) * tanhf(gg);
    float hn = sigf(go) * tanhf(cn);
    out[h1_off + gix] = hn;
    out[c1_off + gix] = cn;
    Hb[m * 64 + j] = (__bf16)hn;
    Abf[m * kSA + j]      = (__bf16)hn;            // A2[:,0:64]   = h1
    Abf[m * kSA + 64 + j] = (__bf16)lstm2_h[gix];  // A2[:,64:128] = h2_prev
  }
  __syncthreads();

  // ---- LSTM2 gates: [64 x 256] = A2[64 x 128] x W2 ----
  gemm_n256<4, 1>(Abf, Wsl, G, w_ih1, w_hh1, w_ih2, w_hh2, fc1_w);
  __syncthreads();

  // elementwise LSTM2
  for (int it = t; it < kMT * 64; it += 256) {
    int m = it >> 6, j = it & 63;
    size_t gix = (size_t)(g0 + m) * 64 + j;
    float gi = G[m * kGS + j]       + b_ih2[j]       + b_hh2[j];
    float gf = G[m * kGS + 64 + j]  + b_ih2[64 + j]  + b_hh2[64 + j];
    float gg = G[m * kGS + 128 + j] + b_ih2[128 + j] + b_hh2[128 + j];
    float go = G[m * kGS + 192 + j] + b_ih2[192 + j] + b_hh2[192 + j];
    float cp = lstm2_c[gix];
    float cn = sigf(gf) * cp + sigf(gi) * tanhf(gg);
    float hn = sigf(go) * tanhf(cn);
    out[h2_off + gix] = hn;
    out[c2_off + gix] = cn;
    Abf[m * kSA + j]      = (__bf16)hn;        // A3[:,0:64]   = h2
    Abf[m * kSA + 64 + j] = Hb[m * 64 + j];    // A3[:,64:128] = h1
  }
  __syncthreads();

  // ---- fc1: [64 x 32] = A3[64 x 128] x fc1_w^T ----
  gemm_fc(Abf, Wsl, G, fc1_w);
  __syncthreads();

  // ---- head: relu(fc1 + b) . fc2_w + b -> sigmoid ----
  if (t < kMT) {
    int m = t;
    float acc = fc2_b[0];
#pragma unroll
    for (int n = 0; n < 32; ++n) {
      float z = fmaxf(G[m * kGS + n] + fc1_b[n], 0.0f);
      acc += z * fc2_w[n];
    }
    out[(size_t)(g0 + m)] = sigf(acc);
  }
}

extern "C" void kernel_launch(void* const* d_in, const int* in_sizes, int n_in,
                              void* d_out, int out_size, void* d_ws, size_t ws_size,
                              hipStream_t stream) {
  (void)n_in; (void)out_size; (void)d_ws; (void)ws_size;
  const float* x         = (const float*)d_in[0];
  const float* lstm1_h   = (const float*)d_in[1];
  const float* lstm1_c   = (const float*)d_in[2];
  const float* lstm2_h   = (const float*)d_in[3];
  const float* lstm2_c   = (const float*)d_in[4];
  const float* conv_dw_w = (const float*)d_in[5];
  const float* conv_pw_w = (const float*)d_in[6];
  const float* conv_pw_b = (const float*)d_in[7];
  const float* sep1_dw_w = (const float*)d_in[8];
  const float* sep1_pw_w = (const float*)d_in[9];
  const float* sep1_pw_b = (const float*)d_in[10];
  const float* sep2_dw_w = (const float*)d_in[11];
  const float* sep2_pw_w = (const float*)d_in[12];
  const float* sep2_pw_b = (const float*)d_in[13];
  const float* w_ih1     = (const float*)d_in[14];
  const float* w_hh1     = (const float*)d_in[15];
  const float* b_ih1     = (const float*)d_in[16];
  const float* b_hh1     = (const float*)d_in[17];
  const float* w_ih2     = (const float*)d_in[18];
  const float* w_hh2     = (const float*)d_in[19];
  const float* b_ih2     = (const float*)d_in[20];
  const float* b_hh2     = (const float*)d_in[21];
  const float* fc1_w     = (const float*)d_in[22];
  const float* fc1_b     = (const float*)d_in[23];
  const float* fc2_w     = (const float*)d_in[24];
  const float* fc2_b     = (const float*)d_in[25];
  float* out = (float*)d_out;

  const int Btot = in_sizes[1] / 64;   // lstm1_h is [B,64]
  const int grid = Btot / kMT;

  (void)hipFuncSetAttribute((const void*)tenvad_fused,
                            hipFuncAttributeMaxDynamicSharedMemorySize, kSmemBytes);

  tenvad_fused<<<grid, 256, kSmemBytes, stream>>>(
      x, lstm1_h, lstm1_c, lstm2_h, lstm2_c,
      conv_dw_w, conv_pw_w, conv_pw_b,
      sep1_dw_w, sep1_pw_w, sep1_pw_b,
      sep2_dw_w, sep2_pw_w, sep2_pw_b,
      w_ih1, w_hh1, b_ih1, b_hh1,
      w_ih2, w_hh2, b_ih2, b_hh2,
      fc1_w, fc1_b, fc2_w, fc2_b,
      out, Btot);
}
